// DecoderAttnRNN_10668698763458
// MI455X (gfx1250) — matrix operand order repaired and checked by reading, not compiled, not run on hardware
//
#include <hip/hip_runtime.h>

#define Bq 64
#define Tq 72
#define Sq 72
#define Eq 128
#define Hq 256
#define Vq 32000

typedef __attribute__((ext_vector_type(16))) _Float16 v16h;
typedef __attribute__((ext_vector_type(8)))  _Float16 v8h;
typedef __attribute__((ext_vector_type(8)))  float    v8f;

// Build WMMA A-operand (16x32 f16, row-major source, lda in halves must keep
// 16B alignment). Lane L holds row (L&15); contiguous K [16h..16h+15]
// where h = L>>4 (matches B operand loads).
__device__ __forceinline__ v16h wmma_a_from(const _Float16* p, int half) {
  v8h a1 = *(const v8h*)(p + 16 * half);
  v8h a2 = *(const v8h*)(p + 16 * half + 8);
  v16h r;
#pragma unroll
  for (int i = 0; i < 8; ++i) { r[i] = a1[i]; r[i + 8] = a2[i]; }
  return r;
}

__device__ __forceinline__ float tanh_c(float x) {
  x = fminf(fmaxf(x, -15.f), 15.f);
  float e = __expf(-2.f * x);
  return (1.f - e) / (1.f + e);
}

// ---------------- phase -1: weight conversion f32 -> f16 ----------------
__global__ void k_cvt(const float* __restrict__ s, _Float16* __restrict__ d, int n) {
  int i = blockIdx.x * blockDim.x + threadIdx.x;
  int stride = gridDim.x * blockDim.x;
  for (; i < n; i += stride) d[i] = (_Float16)s[i];
}

__global__ void k_bias(const float* __restrict__ a, const float* __restrict__ b,
                       float* __restrict__ o, int n) {
  int i = blockIdx.x * blockDim.x + threadIdx.x;
  if (i < n) o[i] = a[i] + b[i];
}

// ---------------- phase 0a: embedding gather (f16 A-matrix for x@W_ih^T) ----
__global__ __launch_bounds__(Eq) void k_gather(const int* __restrict__ tok,
                                               const float* __restrict__ emb,
                                               _Float16* __restrict__ Ax) {
  const int r = blockIdx.x;          // r = b*T + t
  const int e = threadIdx.x;         // 0..127
  const int tk = tok[r];
  Ax[(size_t)r * Eq + e] = (_Float16)emb[(size_t)tk * Eq + e];
}

// ---------------- phase 0b: Xg = Ax @ W_ih^T  (4608 x 1024, K=128) ----------
__global__ __launch_bounds__(256) void k_xw(const _Float16* __restrict__ Ax,
                                            const _Float16* __restrict__ Wih,
                                            float* __restrict__ Xg) {
  const int tid  = threadIdx.x;
  const int wid  = (blockIdx.x << 3) + (tid >> 5);   // global wave-tile id
  const int lane = tid & 31, half = lane >> 4, l15 = lane & 15;
  const int mt = wid >> 6, nt = wid & 63;            // 288 x 64 tiles
  const _Float16* ap = Ax + (size_t)(mt * 16 + l15) * Eq;
  const _Float16* bp = Wih + (size_t)(nt * 16 + l15) * Eq + 16 * half;
  v8f acc = {};
#pragma unroll
  for (int kk = 0; kk < 4; ++kk) {
    v16h av = wmma_a_from(ap + kk * 32, half);
    v16h bv = *(const v16h*)(bp + kk * 32);
    acc = __builtin_amdgcn_wmma_f32_16x16x32_f16(false, av, false, bv, (short)0,
                                                 acc, false, false);
  }
#pragma unroll
  for (int i = 0; i < 8; ++i)
    Xg[(size_t)(mt * 16 + i + 8 * half) * 1024 + nt * 16 + l15] = acc[i];
}

// ---------------- phase 1: sequential LSTM (4 blocks x 16 batches) ----------
__global__ __launch_bounds__(256) void k_lstm(const float* __restrict__ Xg,
                                              const _Float16* __restrict__ Whh,
                                              const float* __restrict__ bias,
                                              const float* __restrict__ h0,
                                              const float* __restrict__ c0,
                                              _Float16* __restrict__ Acat) {
  __shared__ _Float16 hs[16][264];    // h (f16), row pad -> conflict-free A loads
  __shared__ _Float16 gs[16][1040];   // gates staging (f16)
  const int tid = threadIdx.x;
  const int b0 = blockIdx.x << 4;
  for (int idx = tid; idx < 16 * Hq; idx += 256) {
    int bl = idx >> 8, hi = idx & 255;
    hs[bl][hi] = (_Float16)h0[(size_t)(b0 + bl) * Hq + hi];
  }
  const int bl = tid >> 4, hb = tid & 15;
  float c[16];
#pragma unroll
  for (int j = 0; j < 16; ++j) c[j] = c0[(size_t)(b0 + bl) * Hq + hb + 16 * j];
  __syncthreads();
  const int wave = tid >> 5, lane = tid & 31, half = lane >> 4, l15 = lane & 15;

  for (int t = 0; t < Tq; ++t) {
    v8f acc[8];
    // init accumulators with precomputed x@W_ih^T tile
#pragma unroll
    for (int q = 0; q < 8; ++q) {
      const int n0 = ((wave << 3) + q) << 4;
#pragma unroll
      for (int i = 0; i < 8; ++i)
        acc[q][i] = Xg[(size_t)((b0 + i + 8 * half) * Tq + t) * 1024 + n0 + l15];
    }
    // gates += h @ W_hh^T   (K = 256 -> 8 WMMA steps, 64 N-tiles / 8 waves)
#pragma unroll
    for (int kk = 0; kk < 8; ++kk) {
      v16h av = wmma_a_from(&hs[l15][kk * 32], half);
#pragma unroll
      for (int q = 0; q < 8; ++q) {
        const int n0 = ((wave << 3) + q) << 4;
        v16h bv = *(const v16h*)(Whh + (size_t)(n0 + l15) * Hq + kk * 32 + 16 * half);
        acc[q] = __builtin_amdgcn_wmma_f32_16x16x32_f16(false, av, false, bv,
                                                        (short)0, acc[q], false, false);
      }
    }
#pragma unroll
    for (int q = 0; q < 8; ++q) {
      const int n0 = ((wave << 3) + q) << 4;
#pragma unroll
      for (int i = 0; i < 8; ++i)
        gs[i + 8 * half][n0 + l15] = (_Float16)acc[q][i];
    }
    __syncthreads();
    // elementwise LSTM cell; each thread owns 16 (b,hi) cells persistently
#pragma unroll
    for (int j = 0; j < 16; ++j) {
      const int hi = hb + (j << 4);
      float ig = (float)gs[bl][hi]       + bias[hi];
      float fg = (float)gs[bl][256 + hi] + bias[256 + hi];
      float gg = (float)gs[bl][512 + hi] + bias[512 + hi];
      float og = (float)gs[bl][768 + hi] + bias[768 + hi];
      float si = 1.f / (1.f + __expf(-ig));
      float sf = 1.f / (1.f + __expf(-fg));
      float so = 1.f / (1.f + __expf(-og));
      float cn = sf * c[j] + si * tanh_c(gg);
      c[j] = cn;
      float hn = so * tanh_c(cn);
      hs[bl][hi] = (_Float16)hn;
      Acat[(size_t)((b0 + bl) * Tq + t) * 512 + hi] = (_Float16)hn;
    }
    __syncthreads();
  }
}

// ---------------- phase 2: attention for all (b,t) in parallel --------------
__global__ __launch_bounds__(128) void k_attn(_Float16* __restrict__ Acat,
                                              const float* __restrict__ enc,
                                              const int* __restrict__ lens) {
  const int r = blockIdx.x;       // r = b*T + t
  const int b = r / Tq;
  const int tid = threadIdx.x;
  __shared__ float hsh[Hq];
  __shared__ float sc[Sq];
  __shared__ float red[128];
  for (int k = tid; k < Hq; k += 128) hsh[k] = (float)Acat[(size_t)r * 512 + k];
  __syncthreads();
  const int len = lens[b];
  for (int s = tid; s < Sq; s += 128) {
    const float* ep = enc + ((size_t)b * Sq + s) * Hq;
    float d = 0.f;
    for (int k = 0; k < Hq; ++k) d += hsh[k] * ep[k];
    sc[s] = (s < len) ? d * 0.0625f : -1e30f;   // 1/sqrt(256)
  }
  __syncthreads();
  float m = -1e30f;
  for (int s = tid; s < Sq; s += 128) m = fmaxf(m, sc[s]);
  red[tid] = m; __syncthreads();
  for (int off = 64; off > 0; off >>= 1) {
    if (tid < off) red[tid] = fmaxf(red[tid], red[tid + off]);
    __syncthreads();
  }
  m = red[0]; __syncthreads();
  float sum = 0.f;
  for (int s = tid; s < Sq; s += 128) { float e = __expf(sc[s] - m); sc[s] = e; sum += e; }
  red[tid] = sum; __syncthreads();
  for (int off = 64; off > 0; off >>= 1) {
    if (tid < off) red[tid] += red[tid + off];
    __syncthreads();
  }
  const float inv = 1.f / red[0];
  __syncthreads();
  for (int k = tid; k < Hq; k += 128) {
    float a = 0.f;
    for (int s = 0; s < Sq; ++s) a += sc[s] * enc[((size_t)b * Sq + s) * Hq + k];
    Acat[(size_t)r * 512 + Hq + k] = (_Float16)(a * inv);
  }
}

// ---------------- phase 3: logits = Acat @ W_lin^T + b (4608x512x32000) -----
// Block tile 128x256, 512 threads = 16 waves; wave w: m_sub = w&7 (16 rows),
// n group = w>>3 (128 cols = 8 N-tiles). Larger tile halves L2 read traffic
// for the L2-resident W_lin/Acat so reads match the 590MB HBM store floor.
__global__ __launch_bounds__(512) void k_logits(const _Float16* __restrict__ Acat,
                                                const _Float16* __restrict__ Wl,
                                                const float* __restrict__ blin,
                                                float* __restrict__ out) {
  const int tid = threadIdx.x;
  const int w = tid >> 5, lane = tid & 31, half = lane >> 4, l15 = lane & 15;
  const int m_sub = w & 7, n_grp = w >> 3;
  const int row0  = blockIdx.x * 128 + m_sub * 16;
  const int ncol0 = blockIdx.y * 256 + n_grp * 128;
  const _Float16* ap = Acat + (size_t)(row0 + l15) * 512;
  v8f acc[8] = {};
#pragma unroll
  for (int ks = 0; ks < 16; ++ks) {
    const int k0 = ks * 32;
    v16h av = wmma_a_from(ap + k0, half);
#pragma unroll
    for (int j = 0; j < 8; ++j) {
      const _Float16* bp = Wl + (size_t)(ncol0 + j * 16 + l15) * 512 + k0 + 16 * half;
      v16h bv = *(const v16h*)bp;
      acc[j] = __builtin_amdgcn_wmma_f32_16x16x32_f16(false, av, false, bv,
                                                      (short)0, acc[j], false, false);
    }
  }
#pragma unroll
  for (int j = 0; j < 8; ++j) {
    const int n = ncol0 + j * 16 + l15;
    const float bv = blin[n];
#pragma unroll
    for (int i = 0; i < 8; ++i)
      out[(size_t)(row0 + i + 8 * half) * Vq + n] = acc[j][i] + bv;
  }
}

extern "C" void kernel_launch(void* const* d_in, const int* in_sizes, int n_in,
                              void* d_out, int out_size, void* d_ws, size_t ws_size,
                              hipStream_t stream) {
  const int*   tok  = (const int*)  d_in[0];
  const float* enc  = (const float*)d_in[1];
  const int*   lens = (const int*)  d_in[2];
  const float* h0   = (const float*)d_in[3];
  const float* c0   = (const float*)d_in[4];
  const float* emb  = (const float*)d_in[5];
  const float* Wih  = (const float*)d_in[6];
  const float* Whh  = (const float*)d_in[7];
  const float* bih  = (const float*)d_in[8];
  const float* bhh  = (const float*)d_in[9];
  const float* Wlin = (const float*)d_in[10];
  const float* blin = (const float*)d_in[11];
  float* out = (float*)d_out;
  (void)in_sizes; (void)n_in; (void)out_size; (void)ws_size;

  char* ws = (char*)d_ws;
  size_t off = 0;
  auto take = [&](size_t bytes) -> char* {
    char* p = ws + off;
    off += (bytes + 255) & ~(size_t)255;
    return p;
  };
  _Float16* Wlin_h = (_Float16*)take((size_t)Vq * 512 * 2);      // 32.8 MB
  _Float16* Whh_h  = (_Float16*)take((size_t)1024 * Hq * 2);     // 512 KB
  _Float16* Wih_h  = (_Float16*)take((size_t)1024 * Eq * 2);     // 256 KB
  float*    bias   = (float*)   take((size_t)1024 * 4);
  _Float16* Ax     = (_Float16*)take((size_t)Bq * Tq * Eq * 2);  // 1.2 MB
  float*    Xg     = (float*)   take((size_t)Bq * Tq * 1024 * 4);// 18.9 MB
  _Float16* Acat   = (_Float16*)take((size_t)Bq * Tq * 512 * 2); // 4.7 MB

  k_cvt<<<4096, 256, 0, stream>>>(Wlin, Wlin_h, Vq * 512);
  k_cvt<<<512, 256, 0, stream>>>(Whh, Whh_h, 1024 * Hq);
  k_cvt<<<256, 256, 0, stream>>>(Wih, Wih_h, 1024 * Eq);
  k_bias<<<4, 256, 0, stream>>>(bih, bhh, bias, 1024);
  k_gather<<<Bq * Tq, Eq, 0, stream>>>(tok, emb, Ax);
  k_xw<<<(288 * 64) / 8, 256, 0, stream>>>(Ax, Wih_h, Xg);
  k_lstm<<<4, 256, 0, stream>>>(Xg, Whh_h, bias, h0, c0, Acat);
  k_attn<<<Bq * Tq, 128, 0, stream>>>(Acat, enc, lens);
  k_logits<<<dim3((Bq * Tq) / 128, Vq / 256), 512, 0, stream>>>(Acat, Wlin_h, blin, out);
}
